// Set2Set_14448269984511
// MI455X (gfx1250) — compile-verified, matrix operand order checked by reading, not run on hardware
//
#include <hip/hip_runtime.h>
#include <hip/hip_bf16.h>

// ---------------------------------------------------------------------------
// CDNA5 WMMA types (wave32): v_wmma_f32_16x16x32_bf16
// ---------------------------------------------------------------------------
typedef __attribute__((ext_vector_type(16))) __bf16 v16bf;
typedef __attribute__((ext_vector_type(8)))  float  v8f;

#define USE_ASYNC_LDS  1
#define USE_CVT_PK_ASM 1

__device__ __forceinline__ unsigned short f2bf(float f) {
  unsigned int x = __float_as_uint(f);
  unsigned int r = (x + 0x7FFFu + ((x >> 16) & 1u)) >> 16;  // RNE
  return (unsigned short)r;
}
__device__ __forceinline__ float bf2f(unsigned short h) {
  return __uint_as_float(((unsigned int)h) << 16);
}

// packed f32x2 -> bf16x2 in one VALU op
#if defined(__has_builtin) && __has_builtin(__builtin_amdgcn_cvt_pk_bf16_f32)
typedef __attribute__((ext_vector_type(2))) __bf16 v2bf;
__device__ __forceinline__ unsigned int pk_bf16(float lo, float hi) {
  union { v2bf v; unsigned int u; } c;
  c.v = __builtin_amdgcn_cvt_pk_bf16_f32(lo, hi);
  return c.u;
}
#elif USE_CVT_PK_ASM
__device__ __forceinline__ unsigned int pk_bf16(float lo, float hi) {
  unsigned int r;
  asm("v_cvt_pk_bf16_f32 %0, %1, %2" : "=v"(r) : "v"(lo), "v"(hi));
  return r;
}
#else
__device__ __forceinline__ unsigned int pk_bf16(float lo, float hi) {
  return (unsigned int)f2bf(lo) | ((unsigned int)f2bf(hi) << 16);
}
#endif

// DPP lane swaps (no LDS, no waits): quad_perm xor1 / xor2
__device__ __forceinline__ float dpp_xor1_f(float v) {
  return __int_as_float(__builtin_amdgcn_update_dpp(
      0, __float_as_int(v), 0xB1 /*quad_perm[1,0,3,2]*/, 0xF, 0xF, true));
}
__device__ __forceinline__ float dpp_xor2_f(float v) {
  return __int_as_float(__builtin_amdgcn_update_dpp(
      0, __float_as_int(v), 0x4E /*quad_perm[2,3,0,1]*/, 0xF, 0xF, true));
}

// order-preserving float<->uint map for atomicMax on floats
__device__ __forceinline__ unsigned int fmap(float f) {
  unsigned int b = __float_as_uint(f);
  return (b & 0x80000000u) ? ~b : (b | 0x80000000u);
}
__device__ __forceinline__ float funmap(unsigned int u) {
  return (u & 0x80000000u) ? __uint_as_float(u ^ 0x80000000u)
                           : __uint_as_float(~u);
}

// ---------------------------------------------------------------------------
// CDNA5 async global<->LDS (ASYNCcnt). LDS byte offset = low 32 bits of the
// generic pointer (ISA 10.2: aperture sits in the high 32 bits).
// ---------------------------------------------------------------------------
__device__ __forceinline__ void async_g2l_b128(const void* gsrc, void* ldst) {
  unsigned lds_off = (unsigned)(unsigned long long)(ldst);
  asm volatile("global_load_async_to_lds_b128 %0, %1, off"
               :: "v"(lds_off), "v"((unsigned long long)gsrc)
               : "memory");
}
__device__ __forceinline__ void async_l2g_b128(void* gdst, const void* lsrc) {
  unsigned lds_off = (unsigned)(unsigned long long)(lsrc);
  asm volatile("global_store_async_from_lds_b128 %0, %1, off"
               :: "v"((unsigned long long)gdst), "v"(lds_off)
               : "memory");
}
__device__ __forceinline__ void async_wait0() {
  asm volatile("s_wait_asynccnt 0x0" ::: "memory");
}
__device__ __forceinline__ void ds_wait0() {
  asm volatile("s_wait_dscnt 0x0" ::: "memory");
}

union BFrag {
  v16bf v;
  unsigned short u[16];
  uint4 q[2];
};

// A-fragment (16x32 bf16): lane m=lane&15, hi = (lane>=16)?8:0
__device__ __forceinline__ void load_a_frag_f32(BFrag& f, const float* base) {
  float4 a0 = ((const float4*)base)[0];
  float4 a1 = ((const float4*)base)[1];
  float4 a2 = ((const float4*)(base + 16))[0];
  float4 a3 = ((const float4*)(base + 16))[1];
  f.q[0] = make_uint4(pk_bf16(a0.x, a0.y), pk_bf16(a0.z, a0.w),
                      pk_bf16(a1.x, a1.y), pk_bf16(a1.z, a1.w));
  f.q[1] = make_uint4(pk_bf16(a2.x, a2.y), pk_bf16(a2.z, a2.w),
                      pk_bf16(a3.x, a3.y), pk_bf16(a3.z, a3.w));
}

// B-fragment (32x16 bf16) from LDS laid out sB[n*128 + k]
__device__ __forceinline__ void load_b_frag_lds(BFrag& f, const unsigned short* sB,
                                                int ncol, int kb) {
  f.q[0] = *(const uint4*)(sB + ncol * 128 + kb);
  f.q[1] = *(const uint4*)(sB + ncol * 128 + kb + 16);
}

// Double-buffered 8x(16x16) tile-row compute: 32 fragments (f = nt*4 + ks),
// load f+1 into the alternate buffer before consuming f so DS latency hides
// under the previous WMMA (compiler can wait at dscnt<=1 instead of 0).
__device__ __forceinline__ void mma_row_8tiles(v8f acc[8], const BFrag A[4],
                                               const unsigned short* sB,
                                               int lane, int hi) {
  BFrag Bb[2];
  load_b_frag_lds(Bb[0], sB, (lane & 15), hi);  // f=0: nt=0, ks=0
#pragma unroll
  for (int f = 0; f < 32; ++f) {
    const int cu = f & 1;
    if (f < 31) {
      const int fn = f + 1;
      load_b_frag_lds(Bb[cu ^ 1], sB, (fn >> 2) * 16 + (lane & 15),
                      (fn & 3) * 32 + hi);
    }
    const int nt = f >> 2, ks = f & 3;
    acc[nt] = __builtin_amdgcn_wmma_f32_16x16x32_bf16(false, A[ks].v, false, Bb[cu].v,
                                                      (short)0, acc[nt], false, false);
  }
}

// ---------------------------------------------------------------------------
// Kernel 1: xh = x @ lin_w^T + lin_b  -> bf16 [N x 128]
// Persistent blocks: stage lin_w once, loop over 128-row tiles.
// Epilogue: DPP pair-pack (branch-free) -> LDS -> 8x async b128 stores.
// ---------------------------------------------------------------------------
__global__ void xh_gemm(const float* __restrict__ x,
                        const unsigned short* __restrict__ linwbf,
                        const float* __restrict__ linb,
                        unsigned short* __restrict__ xh, int N) {
  __shared__ unsigned short sB[128 * 128];     // 32 KB  B = lin_w (bf16)
  __shared__ unsigned short sOut[8][16 * 128]; // 32 KB  per-wave C staging
  const int tid = threadIdx.x;

#if USE_ASYNC_LDS
  for (int i = tid; i < 2048; i += 256)        // 2048 x 16B = 32 KB
    async_g2l_b128(linwbf + i * 8, sB + i * 8);
  async_wait0();
  __syncthreads();
#else
  for (int i = tid; i < 2048; i += 256)
    ((uint4*)sB)[i] = ((const uint4*)linwbf)[i];
  __syncthreads();
#endif

  const int wave = tid >> 5, lane = tid & 31;
  const int hi = (lane >= 16) ? 8 : 0;
  const int odd = lane & 1;

  float biasv[8];
#pragma unroll
  for (int nt = 0; nt < 8; ++nt) biasv[nt] = linb[nt * 16 + (lane & 15)];

  unsigned int* so32 = (unsigned int*)sOut[wave];
  const int ntiles = (N + 127) / 128;

  for (int tile = blockIdx.x; tile < ntiles; tile += gridDim.x) {
    const int row0 = tile * 128 + wave * 16;
    int arow = row0 + (lane & 15);
    if (arow >= N) arow = N - 1;  // clamp loads

    BFrag A[4];
#pragma unroll
    for (int ks = 0; ks < 4; ++ks)
      load_a_frag_f32(A[ks], x + (size_t)arow * 128 + ks * 32 + hi);

    v8f acc[8];
#pragma unroll
    for (int nt = 0; nt < 8; ++nt) acc[nt] = (v8f){0, 0, 0, 0, 0, 0, 0, 0};

    mma_row_8tiles(acc, A, sB, lane, hi);

    if (row0 + 16 <= N) {          // fast path: branch-free packed epilogue
      async_wait0();               // sOut WAR vs previous tile's async stores
#pragma unroll
      for (int nt = 0; nt < 8; ++nt) {
        const int dw = (nt * 16 + (lane & 15)) >> 1;
#pragma unroll
        for (int r = 0; r < 8; r += 2) {
          const float t0 = acc[nt][r]     + biasv[nt];
          const float t1 = acc[nt][r + 1] + biasv[nt];
          const float s0 = dpp_xor1_f(t0);  // partner column, row r
          const float s1 = dpp_xor1_f(t1);  // partner column, row r+1
          // even lane -> row r cols (c,c+1); odd lane -> row r+1 cols (c-1,c)
          const float lo  = odd ? s1 : t0;
          const float hi2 = odd ? t1 : s0;
          so32[(r + hi + odd) * 64 + dw] = pk_bf16(lo, hi2);
        }
      }
      ds_wait0();  // LDS writes visible before async engine reads them
#pragma unroll
      for (int it = 0; it < 8; ++it) {
        const int u = it * 32 + lane;  // 256 x 16B units (contiguous 4KB)
        async_l2g_b128(xh + (size_t)row0 * 128 + u * 8, sOut[wave] + u * 8);
      }
    } else {  // rare tail tile
#pragma unroll
      for (int nt = 0; nt < 8; ++nt) {
        const int ncol = nt * 16 + (lane & 15);
#pragma unroll
        for (int r = 0; r < 8; ++r) {
          const int m = row0 + r + hi;
          if (m < N) xh[(size_t)m * 128 + ncol] = f2bf(acc[nt][r] + biasv[nt]);
        }
      }
    }
  }
}

// ---------------------------------------------------------------------------
// Kernel 2: gates = [q_star | h] @ [w_ih | w_hh]^T + (b_ih + b_hh)
// M = SEGS, K = 384 (3 async LDS tiles of 128), Nout = 512 (grid.y = 4).
// Epilogue: DPP pair-pack -> f32 LDS (b64 stores) -> async b128 row stores.
// ---------------------------------------------------------------------------
__global__ void gates_gemm(const float* __restrict__ qstar, const float* __restrict__ h,
                           const unsigned short* __restrict__ wcat,
                           const float* __restrict__ bsum,
                           float* __restrict__ gates, int SEGS) {
  __shared__ union {
    unsigned short b[128 * 128];  // 32 KB: B tile while computing
    float f[8 * 16 * 128];        // 64 KB: per-wave f32 C staging afterwards
  } sh;
  const int tid = threadIdx.x;
  const int n0 = blockIdx.y * 128;
  const int wave = tid >> 5, lane = tid & 31;
  const int hi = (lane >= 16) ? 8 : 0;
  const int odd = lane & 1;
  const int row0 = blockIdx.x * 128 + wave * 16;
  int arow = row0 + (lane & 15);
  if (arow >= SEGS) arow = SEGS - 1;

  float biasv[8];
#pragma unroll
  for (int nt = 0; nt < 8; ++nt) biasv[nt] = bsum[n0 + nt * 16 + (lane & 15)];

  v8f acc[8];
#pragma unroll
  for (int nt = 0; nt < 8; ++nt) acc[nt] = (v8f){0, 0, 0, 0, 0, 0, 0, 0};

  for (int kt = 0; kt < 3; ++kt) {
    __syncthreads();
#if USE_ASYNC_LDS
    for (int i = tid; i < 2048; i += 256) {
      const int nl = i >> 4, kc = (i & 15) * 8;
      async_g2l_b128(wcat + (size_t)(n0 + nl) * 384 + kt * 128 + kc,
                     sh.b + nl * 128 + kc);
    }
    async_wait0();
    __syncthreads();
#else
    for (int i = tid; i < 2048; i += 256) {
      const int nl = i >> 4, kc = (i & 15) * 8;
      *(uint4*)(sh.b + nl * 128 + kc) =
          *(const uint4*)(wcat + (size_t)(n0 + nl) * 384 + kt * 128 + kc);
    }
    __syncthreads();
#endif

    const float* Ap; int stride, cb;
    if (kt < 2) { Ap = qstar; stride = 256; cb = kt * 128; }
    else        { Ap = h;     stride = 128; cb = 0; }

    BFrag A[4];
#pragma unroll
    for (int ks = 0; ks < 4; ++ks)
      load_a_frag_f32(A[ks], Ap + (size_t)arow * stride + cb + ks * 32 + hi);

    mma_row_8tiles(acc, A, sh.b, lane, hi);
  }

  __syncthreads();  // everyone done reading sh.b before reuse as f32 staging

  if (row0 + 16 <= SEGS) {
    float* sf = sh.f + wave * 2048;  // 16 rows x 128 cols
#pragma unroll
    for (int nt = 0; nt < 8; ++nt) {
      const int ncl = nt * 16 + (lane & 15);
#pragma unroll
      for (int r = 0; r < 8; r += 2) {
        const float t0 = acc[nt][r]     + biasv[nt];
        const float t1 = acc[nt][r + 1] + biasv[nt];
        const float s0 = dpp_xor1_f(t0);
        const float s1 = dpp_xor1_f(t1);
        const float2 d = odd ? make_float2(s1, t1) : make_float2(t0, s0);
        *(float2*)(sf + (r + hi + odd) * 128 + (ncl & ~1)) = d;
      }
    }
    ds_wait0();
    // each row: 512B contiguous segment gates[(row0+rr)*512 + n0 .. +128)
#pragma unroll
    for (int it = 0; it < 16; ++it) {
      const int cc = lane * 4;  // 32 x 16B units per row
      async_l2g_b128(gates + (size_t)(row0 + it) * 512 + n0 + cc, sf + it * 128 + cc);
    }
  } else {
#pragma unroll
    for (int nt = 0; nt < 8; ++nt) {
      const int gcol = n0 + nt * 16 + (lane & 15);
#pragma unroll
      for (int r = 0; r < 8; ++r) {
        const int m = row0 + r + hi;
        if (m < SEGS) gates[(size_t)m * 512 + gcol] = acc[nt][r] + biasv[nt];
      }
    }
  }
}

// ---------------------------------------------------------------------------
// Pointwise LSTM update, 2 columns per thread (float2 / packed bf16 I/O).
// ---------------------------------------------------------------------------
__global__ void lstm_update(const float* __restrict__ gates, float* __restrict__ h,
                            float* __restrict__ c, float* __restrict__ qstar,
                            unsigned int* __restrict__ qbf, int SEGS) {
  const int idx = blockIdx.x * 256 + threadIdx.x;  // over SEGS*64 col-pairs
  if (idx >= SEGS * 64) return;
  const int s = idx >> 6, jp = idx & 63;
  const size_t gb = (size_t)s * 512 + jp * 2;
  const float2 gi = *(const float2*)(gates + gb);
  const float2 gf = *(const float2*)(gates + gb + 128);
  const float2 gg = *(const float2*)(gates + gb + 256);
  const float2 go = *(const float2*)(gates + gb + 384);
  const float2 cv = *(const float2*)(c + (size_t)s * 128 + jp * 2);

  const float si0 = 1.f / (1.f + __expf(-gi.x)), si1 = 1.f / (1.f + __expf(-gi.y));
  const float sf0 = 1.f / (1.f + __expf(-gf.x)), sf1 = 1.f / (1.f + __expf(-gf.y));
  const float so0 = 1.f / (1.f + __expf(-go.x)), so1 = 1.f / (1.f + __expf(-go.y));
  const float cn0 = sf0 * cv.x + si0 * tanhf(gg.x);
  const float cn1 = sf1 * cv.y + si1 * tanhf(gg.y);
  const float hn0 = so0 * tanhf(cn0);
  const float hn1 = so1 * tanhf(cn1);

  *(float2*)(c + (size_t)s * 128 + jp * 2) = make_float2(cn0, cn1);
  *(float2*)(h + (size_t)s * 128 + jp * 2) = make_float2(hn0, hn1);
  *(float2*)(qstar + (size_t)s * 256 + jp * 2) = make_float2(hn0, hn1);
  *(float2*)(qstar + (size_t)s * 256 + 128 + jp * 2) = make_float2(0.f, 0.f);
  qbf[(size_t)s * 64 + jp] = pk_bf16(hn0, hn1);
}

__global__ void step_init(unsigned int* __restrict__ m_u, float* __restrict__ denom,
                          int SEGS) {
  const int s = blockIdx.x * 256 + threadIdx.x;
  if (s < SEGS) { m_u[s] = fmap(-INFINITY); denom[s] = 0.f; }
}

__global__ void zero_f32(float* __restrict__ p, size_t n) {
  const size_t i = (size_t)blockIdx.x * 256 + threadIdx.x;
  if (i < n) p[i] = 0.f;
}

// ---------------------------------------------------------------------------
// e[n] = dot(xh[n], q[idx[n]]); one wave per row; DPP + bpermute reduce;
// fused segment-max via ordered-uint atomicMax.
// ---------------------------------------------------------------------------
__global__ void e_m_kernel(const unsigned short* __restrict__ xh,
                           const unsigned short* __restrict__ qbf,
                           const int* __restrict__ index,
                           float* __restrict__ e, unsigned int* __restrict__ m_u,
                           int N) {
  const int wave = threadIdx.x >> 5, lane = threadIdx.x & 31;
  const int row = blockIdx.x * 8 + wave;
  if (row >= N) return;
  const int seg = index[row];
  const uint2 xv = *(const uint2*)(xh + (size_t)row * 128 + lane * 4);
  const uint2 qv = *(const uint2*)(qbf + (size_t)seg * 128 + lane * 4);
  float d = 0.f;
  d += bf2f((unsigned short)(xv.x & 0xFFFF)) * bf2f((unsigned short)(qv.x & 0xFFFF));
  d += bf2f((unsigned short)(xv.x >> 16))    * bf2f((unsigned short)(qv.x >> 16));
  d += bf2f((unsigned short)(xv.y & 0xFFFF)) * bf2f((unsigned short)(qv.y & 0xFFFF));
  d += bf2f((unsigned short)(xv.y >> 16))    * bf2f((unsigned short)(qv.y >> 16));
  d += dpp_xor1_f(d);
  d += dpp_xor2_f(d);
  d += __shfl_xor(d, 4, 32);
  d += __shfl_xor(d, 8, 32);
  d += __shfl_xor(d, 16, 32);
  if (lane == 0) {
    e[row] = d;
    atomicMax(m_u + seg, fmap(d));
  }
}

__global__ void ex_denom(const float* __restrict__ e, const int* __restrict__ index,
                         const unsigned int* __restrict__ m_u,
                         float* __restrict__ exb, float* __restrict__ denom, int N) {
  const int i = blockIdx.x * 256 + threadIdx.x;
  if (i >= N) return;
  const int seg = index[i];
  float m = funmap(m_u[seg]);
  if (!isfinite(m)) m = 0.f;
  const float ex = __expf(e[i] - m);
  exb[i] = ex;
  atomicAdd(denom + seg, ex);
}

// ---------------------------------------------------------------------------
// r = segment_sum(a[:,None] * xh): one wave per 128-row run; lane owns 4
// columns (uint2 loads -> coalesced 256B row per wave-instruction).
// Sorted index -> register run-length accumulate, atomic flush on boundary.
// ---------------------------------------------------------------------------
__global__ void r_accum(const unsigned short* __restrict__ xh,
                        const float* __restrict__ exb, const int* __restrict__ index,
                        const float* __restrict__ denom, float* __restrict__ qstar,
                        int N) {
  const int wave = threadIdx.x >> 5, lane = threadIdx.x & 31;
  const int chunk = blockIdx.x * 8 + wave;
  const int r0 = chunk * 128;
  if (r0 >= N) return;
  const int rend = (r0 + 128 < N) ? r0 + 128 : N;

  int cur = -1;
  float inv = 0.f, a0 = 0.f, a1 = 0.f, a2 = 0.f, a3 = 0.f;
  for (int r = r0; r < rend; ++r) {
    const int seg = index[r];
    if (seg != cur) {
      if (cur >= 0) {
        float* dst = &qstar[(size_t)cur * 256 + 128 + lane * 4];
        atomicAdd(dst + 0, a0); atomicAdd(dst + 1, a1);
        atomicAdd(dst + 2, a2); atomicAdd(dst + 3, a3);
      }
      cur = seg; a0 = a1 = a2 = a3 = 0.f;
      inv = 1.f / (denom[seg] + 1e-16f);
    }
    const float a = exb[r] * inv;
    const uint2 xv = *(const uint2*)(xh + (size_t)r * 128 + lane * 4);
    a0 += a * bf2f((unsigned short)(xv.x & 0xFFFF));
    a1 += a * bf2f((unsigned short)(xv.x >> 16));
    a2 += a * bf2f((unsigned short)(xv.y & 0xFFFF));
    a3 += a * bf2f((unsigned short)(xv.y >> 16));
  }
  if (cur >= 0) {
    float* dst = &qstar[(size_t)cur * 256 + 128 + lane * 4];
    atomicAdd(dst + 0, a0); atomicAdd(dst + 1, a1);
    atomicAdd(dst + 2, a2); atomicAdd(dst + 3, a3);
  }
}

// ---------------------------------------------------------------------------
// Weight prep (bf16 packing)
// ---------------------------------------------------------------------------
__global__ void prep_linw(const float* __restrict__ linw,
                          unsigned short* __restrict__ out) {
  const int i = blockIdx.x * 256 + threadIdx.x;  // pair index
  if (i < 128 * 128 / 2)
    *(unsigned int*)(out + i * 2) = pk_bf16(linw[i * 2], linw[i * 2 + 1]);
}

__global__ void prep_wcat(const float* __restrict__ w_ih, const float* __restrict__ w_hh,
                          const float* __restrict__ b_ih, const float* __restrict__ b_hh,
                          unsigned short* __restrict__ wcat, float* __restrict__ bsum) {
  const int g = blockIdx.x;        // 0..511
  const int col = threadIdx.x * 2; // pairs never straddle the 256 boundary
  const float v0 = (col < 256) ? w_ih[(size_t)g * 256 + col]
                               : w_hh[(size_t)g * 128 + (col - 256)];
  const float v1 = (col + 1 < 256) ? w_ih[(size_t)g * 256 + col + 1]
                                   : w_hh[(size_t)g * 128 + (col + 1 - 256)];
  *(unsigned int*)(wcat + (size_t)g * 384 + col) = pk_bf16(v0, v1);
  if (threadIdx.x == 0) bsum[g] = b_ih[g] + b_hh[g];
}

// ---------------------------------------------------------------------------
extern "C" void kernel_launch(void* const* d_in, const int* in_sizes, int n_in,
                              void* d_out, int out_size, void* d_ws, size_t ws_size,
                              hipStream_t stream) {
  const float* x     = (const float*)d_in[0];
  const int*   index = (const int*)d_in[1];
  const float* lin_w = (const float*)d_in[3];
  const float* lin_b = (const float*)d_in[4];
  const float* w_ih  = (const float*)d_in[5];
  const float* w_hh  = (const float*)d_in[6];
  const float* b_ih  = (const float*)d_in[7];
  const float* b_hh  = (const float*)d_in[8];
  float* qstar = (float*)d_out;

  const int HID  = in_sizes[4];            // 128
  const int FIN  = in_sizes[3] / HID;      // 128
  const int N    = in_sizes[0] / FIN;      // 1,000,000
  const int SEGS = out_size / (2 * HID);   // 50,000
  (void)n_in; (void)ws_size;

  char* wsp = (char*)d_ws;
  size_t off = 0;
  auto carve = [&](size_t bytes) -> void* {
    void* p = wsp + off;
    off += (bytes + 255) & ~(size_t)255;
    return p;
  };
  unsigned short* xh     = (unsigned short*)carve((size_t)N * 128 * 2);
  float*          gates  = (float*)carve((size_t)SEGS * 512 * 4);
  float*          hbuf   = (float*)carve((size_t)SEGS * 128 * 4);
  float*          cbuf   = (float*)carve((size_t)SEGS * 128 * 4);
  unsigned int*   qbf    = (unsigned int*)carve((size_t)SEGS * 128 * 2);
  unsigned short* wcat   = (unsigned short*)carve((size_t)512 * 384 * 2);
  unsigned short* linwbf = (unsigned short*)carve((size_t)128 * 128 * 2);
  float*          bsum   = (float*)carve(512 * 4);
  float*          ebuf   = (float*)carve((size_t)N * 4);
  float*          exb    = (float*)carve((size_t)N * 4);
  unsigned int*   m_u    = (unsigned int*)carve((size_t)SEGS * 4);
  float*          denom  = (float*)carve((size_t)SEGS * 4);

  // ---- one-time prep ----
  prep_wcat<<<512, 192, 0, stream>>>(w_ih, w_hh, b_ih, b_hh, wcat, bsum);
  prep_linw<<<32, 256, 0, stream>>>(lin_w, linwbf);

  const size_t nhc = (size_t)SEGS * 128;
  zero_f32<<<(unsigned)((nhc + 255) / 256), 256, 0, stream>>>(hbuf, nhc);
  zero_f32<<<(unsigned)((nhc + 255) / 256), 256, 0, stream>>>(cbuf, nhc);
  const size_t nq = (size_t)SEGS * 256;
  zero_f32<<<(unsigned)((nq + 255) / 256), 256, 0, stream>>>(qstar, nq);

  int xh_blocks = (N + 127) / 128;
  if (xh_blocks > 1024) xh_blocks = 1024;  // persistent
  xh_gemm<<<xh_blocks, 256, 0, stream>>>(x, linwbf, lin_b, xh, N);

  // ---- 3 Set2Set steps ----
  for (int step = 0; step < 3; ++step) {
    dim3 ggrid((SEGS + 127) / 128, 4);
    gates_gemm<<<ggrid, 256, 0, stream>>>(qstar, hbuf, wcat, bsum, gates, SEGS);
    lstm_update<<<(SEGS * 64 + 255) / 256, 256, 0, stream>>>(gates, hbuf, cbuf,
                                                             qstar, qbf, SEGS);
    step_init<<<(SEGS + 255) / 256, 256, 0, stream>>>(m_u, denom, SEGS);
    e_m_kernel<<<(N + 7) / 8, 256, 0, stream>>>(xh, (const unsigned short*)qbf,
                                                index, ebuf, m_u, N);
    ex_denom<<<(N + 255) / 256, 256, 0, stream>>>(ebuf, index, m_u, exb, denom, N);
    r_accum<<<(N + 1023) / 1024, 256, 0, stream>>>(xh, exb, index, denom, qstar, N);
  }
}
